// MLPEncoder_40699110097621
// MI455X (gfx1250) — matrix operand order repaired and checked by reading, not compiled, not run on hardware
//
#include <hip/hip_runtime.h>

// ---------------------------------------------------------------------------
// MLP-Encoder-over-GCN for MI455X (gfx1250, wave32, WMMA).
//
// Pipeline (all on `stream`, all scratch in d_ws):
//   1) deg[n] = 1 + #incoming edges        (atomic f32 adds)
//   2) dinv[n] = rsqrt(deg[n])
//   3) xT[n][b] = data[b][n] * gcn_w ; aggT[n][b] = xT * dinv[n]^2   (LDS transpose)
//   4) edge scatter: aggT[dst][b] += xT[src][b] * dinv[src]*dinv[dst]
//      (transposed layout -> coalesced gathers + coalesced f32 atomics)
//   5) gbf[b][k] = bf16(relu(aggT + gcn_b)), K zero-padded 10000 -> 10016
//   6) W1t[n][k] = bf16(W1[k][n])  (column-major for B-fragment b128 loads)
//   7) GEMM1 (bf16 WMMA 16x16x32, f32 acc): h = relu(g @ W1 + b1) -> bf16
//   8) W2t[n][k] = bf16(W2[k][n])
//   9) GEMM2: out = relu(h @ W2 + b2) -> f32 d_out
// ---------------------------------------------------------------------------

typedef __attribute__((ext_vector_type(8)))  float   v8f;
typedef __attribute__((ext_vector_type(8)))  __bf16  v8bf;
typedef __attribute__((ext_vector_type(16))) __bf16  v16bf;

#define B_   256
#define N_   10000
#define E_   160000
#define H1_  1024
#define H2_  128
#define KP_  10016   // N padded to multiple of 32 for the K loop

// ---------------- degree / normalization --------------------------------
__global__ void k_deg_init(float* deg) {
    int n = blockIdx.x * blockDim.x + threadIdx.x;
    if (n < N_) deg[n] = 1.0f;                     // self-loop
}

__global__ void k_deg_count(const long long* ei, float* deg) {
    int e = blockIdx.x * blockDim.x + threadIdx.x;
    if (e < E_) atomicAdd(&deg[(int)ei[E_ + e]], 1.0f);   // dst = ei[1][e]
}

__global__ void k_dinv(float* deg) {
    int n = blockIdx.x * blockDim.x + threadIdx.x;
    if (n < N_) deg[n] = rsqrtf(deg[n]);
}

// ---------------- xT / aggT build (tiled transpose) ---------------------
// read  data[b][n]  (coalesced over n)
// write xT[n][b], aggT[n][b] = xT * dinv[n]^2  (coalesced over b)
__global__ void k_build_xT(const float* __restrict__ data,
                           const float* __restrict__ gcn_w,
                           const float* __restrict__ dinv,
                           float* __restrict__ xT,
                           float* __restrict__ aggT) {
    __shared__ float tile[32][33];
    const float w = gcn_w[0];
    int n0 = blockIdx.x * 32;
    int b0 = blockIdx.y * 32;
    int tx = threadIdx.x, ty = threadIdx.y;

    int n = n0 + tx, b = b0 + ty;
    if (n < N_) tile[ty][tx] = data[(size_t)b * N_ + n] * w;
    __syncthreads();

    n = n0 + ty; b = b0 + tx;
    if (n < N_) {
        float x  = tile[tx][ty];
        float di = dinv[n];
        xT[(size_t)n * B_ + b]   = x;
        aggT[(size_t)n * B_ + b] = x * di * di;    // self-loop contribution
    }
}

// ---------------- edge scatter ------------------------------------------
// one thread per (edge, batch); 256 consecutive threads share one edge so
// src/dst/dinv loads are wave-uniform and xT/aggT accesses are coalesced.
__global__ void k_edge_scatter(const long long* __restrict__ ei,
                               const float* __restrict__ dinv,
                               const float* __restrict__ xT,
                               float* __restrict__ aggT) {
    unsigned idx = blockIdx.x * 256u + threadIdx.x;
    int e = idx >> 8;           // B_ == 256
    int b = idx & 255;
    int s = (int)ei[e];
    int d = (int)ei[E_ + e];
    float nrm = dinv[s] * dinv[d];
    atomicAdd(&aggT[(size_t)d * B_ + b], xT[(size_t)s * B_ + b] * nrm);
}

// ---------------- g = bf16(relu(agg + gcn_b)), transposed back ----------
// read aggT[n][b] (coalesced over b), write gbf[b][kp=n] (coalesced over n)
__global__ void k_build_g(const float* __restrict__ aggT,
                          const float* __restrict__ gcn_b,
                          __bf16* __restrict__ gbf) {
    __shared__ float tile[32][33];
    const float gb = gcn_b[0];
    int n0 = blockIdx.x * 32;   // over KP_
    int b0 = blockIdx.y * 32;
    int tx = threadIdx.x, ty = threadIdx.y;

    int b = b0 + tx, n = n0 + ty;
    tile[ty][tx] = (n < N_) ? aggT[(size_t)n * B_ + b] : 0.0f;
    __syncthreads();

    n = n0 + tx; b = b0 + ty;
    float v = (n < N_) ? fmaxf(tile[tx][ty] + gb, 0.0f) : 0.0f;
    gbf[(size_t)b * KP_ + n] = (__bf16)v;
}

// ---------------- weight repacks (column-major bf16, K padded) ----------
__global__ void k_build_W1t(const float* __restrict__ W1, __bf16* __restrict__ W1t) {
    __shared__ float tile[32][33];
    int n0 = blockIdx.x * 32;   // over H1_
    int k0 = blockIdx.y * 32;   // over KP_
    int tx = threadIdx.x, ty = threadIdx.y;

    int n = n0 + tx, k = k0 + ty;
    tile[ty][tx] = (k < N_) ? W1[(size_t)k * H1_ + n] : 0.0f;
    __syncthreads();

    n = n0 + ty; k = k0 + tx;
    W1t[(size_t)n * KP_ + k] = (__bf16)tile[tx][ty];
}

__global__ void k_build_W2t(const float* __restrict__ W2, __bf16* __restrict__ W2t) {
    __shared__ float tile[32][33];
    int n0 = blockIdx.x * 32;   // over H2_
    int k0 = blockIdx.y * 32;   // over H1_
    int tx = threadIdx.x, ty = threadIdx.y;

    tile[ty][tx] = W2[(size_t)(k0 + ty) * H2_ + (n0 + tx)];
    __syncthreads();

    W2t[(size_t)(n0 + ty) * H1_ + (k0 + tx)] = (__bf16)tile[tx][ty];
}

// ---------------- WMMA fragment helpers ---------------------------------
// A (16x32 bf16, row-major src, row stride ldk):
//   lane<16 : M=lane,    K 0-15
//   lane>=16: M=lane-16, K 16-31
__device__ inline v16bf load_fragA(const __bf16* base, int kk, int aK, size_t ldk) {
    const v8bf* p = (const v8bf*)(base + kk + 2 * aK);   // base already at row start
    v8bf lo = p[0];          // K 2aK .. 2aK+7
    v8bf hi = p[1];          // K 2aK+8 .. 2aK+15
    return __builtin_shufflevector(lo, hi, 0,1,2,3,4,5,6,7,8,9,10,11,12,13,14,15);
}

// B (32x16 bf16 from column-major W^T, column stride ldk):
//   lane<16 : N=lane,    K 0-15 packed in pairs across v0-7
//   lane>=16: N=lane-16, K 16-31
__device__ inline v16bf load_fragB(const __bf16* colbase, int kk, int bK) {
    const v8bf* p = (const v8bf*)(colbase + kk + bK);
    v8bf lo = p[0];          // K bK .. bK+7
    v8bf hi = p[1];          // K bK+8 .. bK+15
    return __builtin_shufflevector(lo, hi, 0,1,2,3,4,5,6,7,8,9,10,11,12,13,14,15);
}

// ---------------- GEMM1: h = relu(g @ W1 + b1) --------------------------
// M=256 (16 tiles) x N=1024 (64 tiles); each wave computes a 16x64 strip
// (4 accumulators) so the A fragment is reused 4x. 256 waves = 32 blocks.
__global__ __launch_bounds__(256) void k_gemm1(const __bf16* __restrict__ G,
                                               const __bf16* __restrict__ W1t,
                                               const float*  __restrict__ b1,
                                               __bf16* __restrict__ H) {
    int wave = (blockIdx.x * 256 + threadIdx.x) >> 5;   // 0..255
    int lane = threadIdx.x & 31;
    int mt = wave >> 4;            // 0..15
    int ng = wave & 15;            // 0..15 -> n-tiles ng*4 .. ng*4+3
    int lm = lane & 15;
    bool hi = lane >= 16;
    const int aK = hi ? 8 : 0;
    const int bK = hi ? 16 : 0;

    const __bf16* arow = G + (size_t)(mt * 16 + lm) * KP_;
    const __bf16* bcol0 = W1t + (size_t)((ng * 4 + 0) * 16 + lm) * KP_;
    const __bf16* bcol1 = W1t + (size_t)((ng * 4 + 1) * 16 + lm) * KP_;
    const __bf16* bcol2 = W1t + (size_t)((ng * 4 + 2) * 16 + lm) * KP_;
    const __bf16* bcol3 = W1t + (size_t)((ng * 4 + 3) * 16 + lm) * KP_;

    v8f acc0 = {}, acc1 = {}, acc2 = {}, acc3 = {};
    for (int kk = 0; kk < KP_; kk += 32) {
        v16bf A  = load_fragA(arow, kk, aK, KP_);
        v16bf B0 = load_fragB(bcol0, kk, bK);
        v16bf B1 = load_fragB(bcol1, kk, bK);
        v16bf B2 = load_fragB(bcol2, kk, bK);
        v16bf B3 = load_fragB(bcol3, kk, bK);
        acc0 = __builtin_amdgcn_wmma_f32_16x16x32_bf16(false, A, false, B0, (short)0, acc0, false, false);
        acc1 = __builtin_amdgcn_wmma_f32_16x16x32_bf16(false, A, false, B1, (short)0, acc1, false, false);
        acc2 = __builtin_amdgcn_wmma_f32_16x16x32_bf16(false, A, false, B2, (short)0, acc2, false, false);
        acc3 = __builtin_amdgcn_wmma_f32_16x16x32_bf16(false, A, false, B3, (short)0, acc3, false, false);
    }

    // epilogue: +bias, relu, store bf16.  C/D layout: VGPR v -> row v + 8*hi.
    int rbase = mt * 16 + (hi ? 8 : 0);
    v8f accs[4] = {acc0, acc1, acc2, acc3};
#pragma unroll
    for (int j = 0; j < 4; ++j) {
        int col = (ng * 4 + j) * 16 + lm;
        float bias = b1[col];
#pragma unroll
        for (int v = 0; v < 8; ++v) {
            float val = fmaxf(accs[j][v] + bias, 0.0f);
            H[(size_t)(rbase + v) * H1_ + col] = (__bf16)val;
        }
    }
}

// ---------------- GEMM2: out = relu(h @ W2 + b2) ------------------------
// M=256 (16 tiles) x N=128 (8 tiles) = 128 waves = 16 blocks.
__global__ __launch_bounds__(256) void k_gemm2(const __bf16* __restrict__ H,
                                               const __bf16* __restrict__ W2t,
                                               const float*  __restrict__ b2,
                                               float* __restrict__ out) {
    int wave = (blockIdx.x * 256 + threadIdx.x) >> 5;   // 0..127
    int lane = threadIdx.x & 31;
    int mt = wave >> 3;            // 0..15
    int nt = wave & 7;             // 0..7
    int lm = lane & 15;
    bool hi = lane >= 16;
    const int aK = hi ? 8 : 0;
    const int bK = hi ? 16 : 0;

    const __bf16* arow = H + (size_t)(mt * 16 + lm) * H1_;
    const __bf16* bcol = W2t + (size_t)(nt * 16 + lm) * H1_;

    v8f acc = {};
    for (int kk = 0; kk < H1_; kk += 32) {
        v16bf A = load_fragA(arow, kk, aK, H1_);
        v16bf B = load_fragB(bcol, kk, bK);
        acc = __builtin_amdgcn_wmma_f32_16x16x32_bf16(false, A, false, B, (short)0, acc, false, false);
    }

    int rbase = mt * 16 + (hi ? 8 : 0);
    int col = nt * 16 + lm;
    float bias = b2[col];
#pragma unroll
    for (int v = 0; v < 8; ++v)
        out[(size_t)(rbase + v) * H2_ + col] = fmaxf(acc[v] + bias, 0.0f);
}

// ---------------- launch -------------------------------------------------
extern "C" void kernel_launch(void* const* d_in, const int* in_sizes, int n_in,
                              void* d_out, int out_size, void* d_ws, size_t ws_size,
                              hipStream_t stream) {
    const float*      data  = (const float*)d_in[0];   // [B,N]
    const float*      gcn_w = (const float*)d_in[1];   // [1]
    const float*      gcn_b = (const float*)d_in[2];   // [1]
    const float*      W1    = (const float*)d_in[3];   // [N,H1]
    const float*      b1    = (const float*)d_in[4];   // [H1]
    const float*      W2    = (const float*)d_in[5];   // [H1,H2]
    const float*      b2    = (const float*)d_in[6];   // [H2]
    const long long*  ei    = (const long long*)d_in[7]; // [2,E] int64
    float* out = (float*)d_out;                        // [B,H2]

    // workspace carve-up (256B aligned), ~47 MB total
    char* ws = (char*)d_ws;
    size_t off = 0;
    auto carve = [&](size_t bytes) {
        char* p = ws + off;
        off = (off + bytes + 255) & ~(size_t)255;
        return p;
    };
    float*  deg  = (float*)carve((size_t)N_ * 4);              // becomes dinv
    float*  xT   = (float*)carve((size_t)N_ * B_ * 4);
    float*  aggT = (float*)carve((size_t)N_ * B_ * 4);
    __bf16* gbf  = (__bf16*)carve((size_t)B_ * KP_ * 2);
    __bf16* W1t  = (__bf16*)carve((size_t)H1_ * KP_ * 2);
    __bf16* hbf  = (__bf16*)carve((size_t)B_ * H1_ * 2);
    __bf16* W2t  = (__bf16*)carve((size_t)H2_ * H1_ * 2);
    (void)ws_size;

    dim3 t32(32, 32);

    k_deg_init    <<<(N_ + 255) / 256, 256, 0, stream>>>(deg);
    k_deg_count   <<<(E_ + 255) / 256, 256, 0, stream>>>(ei, deg);
    k_dinv        <<<(N_ + 255) / 256, 256, 0, stream>>>(deg);
    k_build_xT    <<<dim3((N_ + 31) / 32, B_ / 32), t32, 0, stream>>>(data, gcn_w, deg, xT, aggT);
    k_edge_scatter<<<(E_ * B_) / 256, 256, 0, stream>>>(ei, deg, xT, aggT);
    k_build_g     <<<dim3(KP_ / 32, B_ / 32), t32, 0, stream>>>(aggT, gcn_b, gbf);
    k_build_W1t   <<<dim3(H1_ / 32, KP_ / 32), t32, 0, stream>>>(W1, W1t);
    k_build_W2t   <<<dim3(H2_ / 32, H1_ / 32), t32, 0, stream>>>(W2, W2t);
    k_gemm1       <<<32, 256, 0, stream>>>(gbf, W1t, b1, hbf);
    k_gemm2       <<<16, 256, 0, stream>>>(hbf, W2t, b2, out);
}
